// ParallelBlock_30382598652249
// MI455X (gfx1250) — matrix operand order repaired and checked by reading, not compiled, not run on hardware
//
#include <hip/hip_runtime.h>
#include <stdint.h>

#define HDIM 2560
#define H3   7680
#define H4   10240
#define NHEADS 32
#define DHEAD  80
#define DPAD   96
#define SEQ    2048
#define BATCH  2
#define ROWS   (BATCH * SEQ) // 4096
#define AWAVES 4
#define NT     8   // n-tiles (of 16) per GEMM wave => 16x128 strip

typedef __attribute__((ext_vector_type(16))) __bf16 v16bf;
typedef __attribute__((ext_vector_type(8)))  float  v8f;

// round-to-nearest-even f32 -> bf16 (raw u16; no __bf16 scalar math needed)
static __device__ __forceinline__ unsigned short f2bf(float f) {
  union { float f; unsigned int u; } c; c.f = f;
  unsigned int u = c.u;
  return (unsigned short)((u + 0x7FFFu + ((u >> 16) & 1u)) >> 16);
}

// ---------------- weight transpose + fp32->bf16 convert: W[K][N] -> Wt[N][K] ----------------
__global__ void convT(const float* __restrict__ W, unsigned short* __restrict__ Wt, int K, int N) {
  __shared__ float tile[32][33];
  int nb = blockIdx.x * 32, kb = blockIdx.y * 32;
  int tx = threadIdx.x & 31, ty = threadIdx.x >> 5;
  for (int r = ty; r < 32; r += 8)
    tile[r][tx] = W[(size_t)(kb + r) * N + nb + tx];
  __syncthreads();
  for (int r = ty; r < 32; r += 8)
    Wt[(size_t)(nb + r) * K + kb + tx] = f2bf(tile[tx][r]);
}

// ---------------- layernorm -> bf16 ----------------
__global__ void ln_kernel(const float* __restrict__ x, const float* __restrict__ w,
                          const float* __restrict__ b, unsigned short* __restrict__ out) {
  int row = blockIdx.x;
  const float* xr = x + (size_t)row * HDIM;
  int tid = threadIdx.x;
  float s = 0.f, sq = 0.f;
  for (int i = tid; i < HDIM; i += 256) { float v = xr[i]; s += v; sq += v * v; }
  for (int off = 16; off > 0; off >>= 1) { s += __shfl_xor(s, off, 32); sq += __shfl_xor(sq, off, 32); }
  __shared__ float red[16];
  int wave = tid >> 5, lane = tid & 31;
  if (lane == 0) { red[wave] = s; red[8 + wave] = sq; }
  __syncthreads();
  float ts = 0.f, tq = 0.f;
  for (int wv = 0; wv < 8; ++wv) { ts += red[wv]; tq += red[8 + wv]; }
  float mean = ts * (1.0f / HDIM);
  float var  = tq * (1.0f / HDIM) - mean * mean;
  float rstd = rsqrtf(var + 1e-5f);
  unsigned short* o = out + (size_t)row * HDIM;
  for (int i = tid; i < HDIM; i += 256)
    o[i] = f2bf((xr[i] - mean) * rstd * w[i] + b[i]);
}

// ---------------- generic WMMA GEMM: C[M][N] = A[M][K](bf16) * Wt[N][K]^T + bias ----------------
// Each wave: 16x128 strip (8 x 16x16 accumulators, one A fragment reused 8x per k-step).
// EPI=0: fp32 output.  EPI=1: fast-gelu epilogue, bf16 output.
template <int EPI>
__global__ __launch_bounds__(256, 1) void gemm_wmma(const unsigned short* __restrict__ A,
                                                    const unsigned short* __restrict__ Wt,
                                                    const float* __restrict__ bias,
                                                    void* __restrict__ Cout, int M, int N, int K) {
  const int lane = threadIdx.x & 31;
  const int wave = threadIdx.x >> 5;
  const int nl   = lane & 15;
  const int selA = (lane < 16) ? 0 : 16;  // A k order used here: upper half-wave holds K 16..31
  const int selB = (lane < 16) ? 0 : 16;  // B k order used here: upper half-wave holds K 16..31
  const int ntile0 = blockIdx.x * NT;
  const int mtile  = blockIdx.y * 8 + wave;

  const unsigned int*   ap    = (const unsigned int*)(A + (size_t)(mtile * 16 + nl) * K + selA);
  const unsigned short* wbase = Wt + (size_t)(ntile0 * 16 + nl) * K + selB;
  const size_t wstride = (size_t)16 * K;

  union AF { v16bf v; unsigned int u[8]; };
  v8f acc[NT];
#pragma unroll
  for (int t = 0; t < NT; ++t) acc[t] = (v8f){};

  for (int k = 0; k < K; k += 32) {
    const unsigned int* a = ap + (k >> 1);
    __builtin_prefetch(a + 32, 0, 0);      // global_prefetch_b8 of A two k-steps ahead
    AF af;
#pragma unroll
    for (int j = 0; j < 4; ++j) { af.u[j] = a[j]; af.u[j + 4] = a[j + 4]; }
#pragma unroll
    for (int t = 0; t < NT; ++t) {
      v16bf bt = *(const v16bf*)(wbase + (size_t)t * wstride + k);
      acc[t] = __builtin_amdgcn_wmma_f32_16x16x32_bf16(false, af.v, false, bt, (short)0, acc[t], false, false);
    }
  }

  const int half8 = (lane < 16) ? 0 : 8;
#pragma unroll
  for (int t = 0; t < NT; ++t) {
    int n = (ntile0 + t) * 16 + nl;
    float bv = bias[n];
#pragma unroll
    for (int r = 0; r < 8; ++r) {
      size_t row = (size_t)mtile * 16 + half8 + r;
      float v = acc[t][r] + bv;
      if (EPI == 1) {
        float g = 0.5f * v * (1.0f + tanhf(0.7978845608f * (v + 0.044715f * v * v * v)));
        ((unsigned short*)Cout)[row * N + n] = f2bf(g);
      } else {
        ((float*)Cout)[row * N + n] = v;
      }
    }
  }
}

// ---------------- RoPE + pack q/k into [b,h,s,96] bf16, V transposed [b,h,d,s] bf16 ----------------
__global__ void rope_pack(const float* __restrict__ qkv, unsigned short* __restrict__ qp,
                          unsigned short* __restrict__ kp, unsigned short* __restrict__ vt) {
  int idx = blockIdx.x * blockDim.x + threadIdx.x;
  if (idx >= BATCH * NHEADS * SEQ) return;
  int s  = idx % SEQ;
  int bh = idx / SEQ;
  int h  = bh % NHEADS;
  int b  = bh / NHEADS;
  size_t row = (size_t)b * SEQ + s;
  const float* q = qkv + row * H3 + h * DHEAD;
  const float* k = q + HDIM;
  const float* v = q + 2 * HDIM;
  unsigned short* qo = qp + ((size_t)bh * SEQ + s) * DPAD;
  unsigned short* ko = kp + ((size_t)bh * SEQ + s) * DPAD;
  unsigned short* vo = vt + (size_t)bh * DHEAD * SEQ + s;
  float fs = (float)s;
  const float lf = 0.5756462732485114f; // ln(10000)/16
#pragma unroll
  for (int d = 0; d < 16; ++d) {
    float inv = __expf(-(float)d * lf);
    float ang = fs * inv;
    float c = __cosf(ang), sn = __sinf(ang);
    float q1 = q[d], q2 = q[d + 16];
    float k1 = k[d], k2 = k[d + 16];
    qo[d]      = f2bf(q1 * c - q2 * sn);
    qo[d + 16] = f2bf(q1 * sn + q2 * c);
    ko[d]      = f2bf(k1 * c - k2 * sn);
    ko[d + 16] = f2bf(k1 * sn + k2 * c);
  }
  for (int d = 32; d < DHEAD; ++d) { qo[d] = f2bf(q[d]); ko[d] = f2bf(k[d]); }
  for (int d = DHEAD; d < DPAD; ++d) { qo[d] = 0; ko[d] = 0; }
  for (int d = 0; d < DHEAD; ++d) vo[(size_t)d * SEQ] = f2bf(v[d]);
}

// ---------------- causal flash attention, one 16-query tile per wave ----------------
__global__ __launch_bounds__(32 * AWAVES, 1) void attn_kernel(const unsigned short* __restrict__ qp,
                                                              const unsigned short* __restrict__ kp,
                                                              const unsigned short* __restrict__ vt,
                                                              unsigned short* __restrict__ attn) {
  const int lane = threadIdx.x & 31;
  const int wave = threadIdx.x >> 5;
  const int tile = blockIdx.x * AWAVES + wave;
  const int qtiles = SEQ / 16;
  const int qt = tile % qtiles;
  const int bh = tile / qtiles;
  const int h = bh % NHEADS;
  const int b = bh / NHEADS;
  const int q0 = qt * 16;
  const int nl = lane & 15;
  const int selA = (lane < 16) ? 0 : 16;
  const int selB = (lane < 16) ? 0 : 16;
  const int half8 = (lane < 16) ? 0 : 8;

  const unsigned short* Qb = qp + (size_t)bh * SEQ * DPAD;
  const unsigned short* Kb = kp + (size_t)bh * SEQ * DPAD;
  const unsigned short* Vb = vt + (size_t)bh * DHEAD * SEQ;

  union AF { v16bf v; unsigned int u[8]; };
  AF qf[3];
  {
    const unsigned int* qpu = (const unsigned int*)(Qb + (size_t)(q0 + nl) * DPAD);
#pragma unroll
    for (int c = 0; c < 3; ++c) {
      const unsigned int* a = qpu + ((c * 32 + selA) >> 1);
#pragma unroll
      for (int j = 0; j < 4; ++j) { qf[c].u[j] = a[j]; qf[c].u[j + 4] = a[j + 4]; }
    }
  }

  v8f o0 = {}, o1 = {}, o2 = {}, o3 = {}, o4 = {};
  float mrow[8], lrow[8];
#pragma unroll
  for (int r = 0; r < 8; ++r) { mrow[r] = -1e30f; lrow[r] = 0.f; }

  __shared__ unsigned short pbuf[AWAVES][16 * 32];
  unsigned short* pb = pbuf[wave];

  const float scale = 0.11180339887498948f; // 1/sqrt(80)
  const int kbmax = (q0 + 15) >> 5;
  for (int kb = 0; kb <= kbmax; ++kb) {
    const int key0 = kb * 32;
    v8f s0 = {}, s1 = {};
    const unsigned short* K0 = Kb + (size_t)(key0 + nl) * DPAD + selB;
    const unsigned short* K1 = K0 + 16 * DPAD;
#pragma unroll
    for (int c = 0; c < 3; ++c) {
      v16bf b0 = *(const v16bf*)(K0 + c * 32);
      v16bf b1 = *(const v16bf*)(K1 + c * 32);
      s0 = __builtin_amdgcn_wmma_f32_16x16x32_bf16(false, qf[c].v, false, b0, (short)0, s0, false, false);
      s1 = __builtin_amdgcn_wmma_f32_16x16x32_bf16(false, qf[c].v, false, b1, (short)0, s1, false, false);
    }
    const int keyA = key0 + nl, keyB = keyA + 16;
    float alpha[8];
#pragma unroll
    for (int r = 0; r < 8; ++r) {
      int qg = q0 + half8 + r;
      float v0 = (keyA <= qg) ? s0[r] * scale : -1e30f;
      float v1 = (keyB <= qg) ? s1[r] * scale : -1e30f;
      float mx = fmaxf(v0, v1);
      for (int off = 1; off < 16; off <<= 1) mx = fmaxf(mx, __shfl_xor(mx, off, 32));
      float mnew = fmaxf(mrow[r], mx);
      float a  = __expf(mrow[r] - mnew);
      float p0 = __expf(v0 - mnew);
      float p1 = __expf(v1 - mnew);
      float sum = p0 + p1;
      for (int off = 1; off < 16; off <<= 1) sum += __shfl_xor(sum, off, 32);
      lrow[r] = lrow[r] * a + sum;
      mrow[r] = mnew;
      alpha[r] = a;
      pb[(half8 + r) * 32 + nl]      = f2bf(p0); // C-layout -> LDS
      pb[(half8 + r) * 32 + 16 + nl] = f2bf(p1);
    }
#pragma unroll
    for (int r = 0; r < 8; ++r) {
      o0[r] *= alpha[r]; o1[r] *= alpha[r]; o2[r] *= alpha[r]; o3[r] *= alpha[r]; o4[r] *= alpha[r];
    }
    asm volatile("s_wait_dscnt 0" ::: "memory"); // P tile visible before A-fragment reload
    AF pf;
    {
      const unsigned int* lp = (const unsigned int*)(pb + nl * 32 + selA);
#pragma unroll
      for (int j = 0; j < 4; ++j) { pf.u[j] = lp[j]; pf.u[j + 4] = lp[j + 4]; }
    }
    const unsigned short* V0 = Vb + (size_t)nl * SEQ + key0 + selB;
    v16bf vb0 = *(const v16bf*)(V0);
    v16bf vb1 = *(const v16bf*)(V0 + 16 * SEQ);
    v16bf vb2 = *(const v16bf*)(V0 + 32 * SEQ);
    v16bf vb3 = *(const v16bf*)(V0 + 48 * SEQ);
    v16bf vb4 = *(const v16bf*)(V0 + 64 * SEQ);
    o0 = __builtin_amdgcn_wmma_f32_16x16x32_bf16(false, pf.v, false, vb0, (short)0, o0, false, false);
    o1 = __builtin_amdgcn_wmma_f32_16x16x32_bf16(false, pf.v, false, vb1, (short)0, o1, false, false);
    o2 = __builtin_amdgcn_wmma_f32_16x16x32_bf16(false, pf.v, false, vb2, (short)0, o2, false, false);
    o3 = __builtin_amdgcn_wmma_f32_16x16x32_bf16(false, pf.v, false, vb3, (short)0, o3, false, false);
    o4 = __builtin_amdgcn_wmma_f32_16x16x32_bf16(false, pf.v, false, vb4, (short)0, o4, false, false);
  }

#pragma unroll
  for (int r = 0; r < 8; ++r) {
    float inv = 1.0f / lrow[r];
    size_t rowg = (size_t)b * SEQ + q0 + half8 + r;
    unsigned short* outp = attn + rowg * HDIM + h * DHEAD + nl;
    outp[0]  = f2bf(o0[r] * inv);
    outp[16] = f2bf(o1[r] * inv);
    outp[32] = f2bf(o2[r] * inv);
    outp[48] = f2bf(o3[r] * inv);
    outp[64] = f2bf(o4[r] * inv);
  }
}

// ---------------- residual sum ----------------
__global__ void final_add(const float* __restrict__ x, const float* __restrict__ a,
                          const float* __restrict__ m, float* __restrict__ o, int n) {
  int i = blockIdx.x * 256 + threadIdx.x;
  if (i < n) o[i] = x[i] + a[i] + m[i];
}

extern "C" void kernel_launch(void* const* d_in, const int* in_sizes, int n_in,
                              void* d_out, int out_size, void* d_ws, size_t ws_size,
                              hipStream_t stream) {
  (void)in_sizes; (void)n_in; (void)out_size; (void)ws_size;
  const float* hidden = (const float*)d_in[0];
  const float* ln_w   = (const float*)d_in[1];
  const float* ln_b   = (const float*)d_in[2];
  const float* qkv_w  = (const float*)d_in[3];
  const float* qkv_b  = (const float*)d_in[4];
  const float* out_w  = (const float*)d_in[5];
  const float* out_b  = (const float*)d_in[6];
  const float* fc1_w  = (const float*)d_in[7];
  const float* fc1_b  = (const float*)d_in[8];
  const float* fc2_w  = (const float*)d_in[9];
  const float* fc2_b  = (const float*)d_in[10];

  char* ws = (char*)d_ws;
  size_t off = 0;
  auto take = [&](size_t bytes) { char* p = ws + off; off = (off + bytes + 255) & ~(size_t)255; return p; };

  unsigned short* qkvt = (unsigned short*)take((size_t)H3 * HDIM * 2);
  unsigned short* outt = (unsigned short*)take((size_t)HDIM * HDIM * 2);
  unsigned short* fc1t = (unsigned short*)take((size_t)H4 * HDIM * 2);
  unsigned short* fc2t = (unsigned short*)take((size_t)HDIM * H4 * 2);
  unsigned short* lnb  = (unsigned short*)take((size_t)ROWS * HDIM * 2);
  float*          qkv  = (float*)take((size_t)ROWS * H3 * 4);
  unsigned short* qp   = (unsigned short*)take((size_t)BATCH * NHEADS * SEQ * DPAD * 2);
  unsigned short* kp   = (unsigned short*)take((size_t)BATCH * NHEADS * SEQ * DPAD * 2);
  unsigned short* vt   = (unsigned short*)take((size_t)BATCH * NHEADS * DHEAD * SEQ * 2);
  unsigned short* attn = (unsigned short*)take((size_t)ROWS * HDIM * 2);
  float*          aout = (float*)take((size_t)ROWS * HDIM * 4);
  unsigned short* mlph = (unsigned short*)take((size_t)ROWS * H4 * 2);
  float*          mlp  = (float*)take((size_t)ROWS * HDIM * 4);

  convT<<<dim3(H3 / 32, HDIM / 32), 256, 0, stream>>>(qkv_w, qkvt, HDIM, H3);
  convT<<<dim3(HDIM / 32, HDIM / 32), 256, 0, stream>>>(out_w, outt, HDIM, HDIM);
  convT<<<dim3(H4 / 32, HDIM / 32), 256, 0, stream>>>(fc1_w, fc1t, HDIM, H4);
  convT<<<dim3(HDIM / 32, H4 / 32), 256, 0, stream>>>(fc2_w, fc2t, H4, HDIM);

  ln_kernel<<<ROWS, 256, 0, stream>>>(hidden, ln_w, ln_b, lnb);

  gemm_wmma<0><<<dim3(H3 / (16 * NT), ROWS / 128), 256, 0, stream>>>(lnb, qkvt, qkv_b, qkv, ROWS, H3, HDIM);

  rope_pack<<<(BATCH * NHEADS * SEQ) / 256, 256, 0, stream>>>(qkv, qp, kp, vt);

  attn_kernel<<<(BATCH * NHEADS * (SEQ / 16)) / AWAVES, 32 * AWAVES, 0, stream>>>(qp, kp, vt, attn);

  gemm_wmma<0><<<dim3(HDIM / (16 * NT), ROWS / 128), 256, 0, stream>>>(attn, outt, out_b, aout, ROWS, HDIM, HDIM);
  gemm_wmma<1><<<dim3(H4 / (16 * NT), ROWS / 128), 256, 0, stream>>>(lnb, fc1t, fc1_b, mlph, ROWS, H4, HDIM);
  gemm_wmma<0><<<dim3(HDIM / (16 * NT), ROWS / 128), 256, 0, stream>>>(mlph, fc2t, fc2_b, mlp, ROWS, HDIM, H4);

  final_add<<<(ROWS * HDIM) / 256, 256, 0, stream>>>(hidden, aout, mlp, (float*)d_out, ROWS * HDIM);
}
